// GCNEncoder_9216999817889
// MI455X (gfx1250) — matrix operand order repaired and checked by reading, not compiled, not run on hardware
//
#include <hip/hip_runtime.h>
#include <stdint.h>

// ---------------------------------------------------------------------------
// GCN encoder for MI455X (gfx1250, wave32).
//   gemm (bf16 WMMA, double-buffered async-to-LDS B staging) -> CSR
//   gather+ReLU -> gemm -> gather+ReLU -> fp32 streaming projection
//   (HBM roofline: Wm/Wv 736 MB read once ~= 32 us at 23.3 TB/s; GEMMs on
//   v_wmma_f32_16x16x32_bf16; all activations L2-resident in the 192 MB L2).
// ---------------------------------------------------------------------------

#define NB     8
#define NN     2810
#define D0     512
#define D1     1024
#define D2     256
#define NE     89920
#define MROWS  (NB * NN)        // 22480
#define KFLAT  (NN * D2)        // 719360

typedef __attribute__((ext_vector_type(16))) __bf16 v16bf;
typedef __attribute__((ext_vector_type(8)))  float  v8f;

static __device__ __forceinline__ unsigned short f2bf(float f) {
    union { float f; uint32_t u; } x; x.f = f;
    uint32_t u = x.u;
    uint32_t r = u + 0x7FFFu + ((u >> 16) & 1u);      // round-to-nearest-even
    if ((u & 0x7F800000u) == 0x7F800000u) r = u;      // inf/nan: truncate
    return (unsigned short)(r >> 16);
}

// --------------------------- fp32 -> bf16 ----------------------------------
__global__ void f32_to_bf16_kernel(const float* __restrict__ in,
                                   unsigned short* __restrict__ out, int n) {
    int i = blockIdx.x * blockDim.x + threadIdx.x;
    int stride = gridDim.x * blockDim.x;
    for (; i < n; i += stride) out[i] = f2bf(in[i]);
}

// --------------------------- CSR construction ------------------------------
__global__ void zero_i32_kernel(int* p, int n) {
    int i = blockIdx.x * blockDim.x + threadIdx.x;
    if (i < n) p[i] = 0;
}

__global__ void deg_kernel(const int* __restrict__ ei, int* __restrict__ deg) {
    int e = blockIdx.x * blockDim.x + threadIdx.x;
    if (e < NE) atomicAdd(&deg[ei[e]], 1);            // dst = edge_index[0][e]
}

__global__ __launch_bounds__(256) void scan_kernel(const int* __restrict__ deg,
                                                   int* __restrict__ rowptr) {
    __shared__ int sums[256];
    const int tid = threadIdx.x;
    const int per = (NN + 255) / 256;                 // 11
    const int start = tid * per;
    int local = 0;
    for (int i = 0; i < per; ++i) {
        int idx = start + i;
        if (idx < NN) local += deg[idx];
    }
    sums[tid] = local;
    __syncthreads();
    for (int off = 1; off < 256; off <<= 1) {         // inclusive scan
        int v = sums[tid];
        int add = (tid >= off) ? sums[tid - off] : 0;
        __syncthreads();
        sums[tid] = v + add;
        __syncthreads();
    }
    int run = (tid == 0) ? 0 : sums[tid - 1];         // exclusive offset
    for (int i = 0; i < per; ++i) {
        int idx = start + i;
        if (idx < NN) { rowptr[idx] = run; run += deg[idx]; }
    }
    if (tid == 255) rowptr[NN] = run;                 // total = NE
}

__global__ void fill_kernel(const int* __restrict__ ei, const float* __restrict__ ew,
                            const int* __restrict__ rowptr, int* __restrict__ cursor,
                            int* __restrict__ esrc, float* __restrict__ ewo) {
    int e = blockIdx.x * blockDim.x + threadIdx.x;
    if (e >= NE) return;
    int d = ei[e];
    int s = ei[NE + e];
    int p = rowptr[d] + atomicAdd(&cursor[d], 1);
    esrc[p] = s;
    ewo[p]  = ew[e];
}

// ------------- bf16 GEMM: C[M,N] = A[M,K] * B[N,K]^T (fp32 out) ------------
// Workgroup = 8 waves, 128(M) x 64(N) tile. The 64x32 bf16 B-tile per K-step
// is staged into double-buffered LDS with async global->LDS DMA: the DMA for
// step k+1 runs under the 4 back-to-back v_wmma_f32_16x16x32_bf16 of step k
// (s_wait_asynccnt 1 keeps one transfer in flight). The last K-step is peeled
// so the steady-state loop body is branch-free, and a sched_barrier pins all
// fragment loads ahead of the WMMA group so the 4 WMMAs issue back-to-back.
__global__ __launch_bounds__(256) void gemm_bf16_kernel(
    const unsigned short* __restrict__ A, const unsigned short* __restrict__ B,
    float* __restrict__ C, int M, int N, int K) {
    __shared__ unsigned short ldsB[2][64 * 32];       // 2 x 4 KB
    const int lane = threadIdx.x & 31;
    const int wave = threadIdx.x >> 5;
    int m0 = blockIdx.x * 128 + wave * 16;
    if (m0 > M - 16) m0 = M - 16;                     // clamp: duplicate, valid
    const int n0 = blockIdx.y * 64;

    // A fragment addressing (contiguous K order, matching B):
    //   lanes 0-15: row M=l, K = kk+{0..15}
    //   lanes 16-31: row M=l-16, K = kk+{16..31}
    const int arow  = m0 + (lane & 15);
    const int khalf = (lane >> 4) * 16;
    const unsigned short* Arow = A + (size_t)arow * K + khalf;

    // B staging: 256 threads x 16B == 64 rows x 64B (== 32 bf16) exactly.
    const int sr = threadIdx.x >> 2;                  // 0..63 (row of B tile)
    const int sq = threadIdx.x & 3;                   // 16B quarter within row
    const unsigned short* Bstage = B + (size_t)(n0 + sr) * K + sq * 8;
    const uint32_t ldsBase = (uint32_t)(uintptr_t)(&ldsB[0][0]);
    const uint32_t ldsDst  = ldsBase + (uint32_t)(sr * 64 + sq * 16);

    union F16 { v16bf v; uint4 u[2]; };
    v8f acc[4] = {};

    // per-step compute: load A frag + all 4 B frags, fence the scheduler,
    // then issue the 4 WMMAs as one group.
    auto compute = [&](int kk, int buf) {
        F16 af;
        const unsigned short* ap = Arow + kk;
        af.u[0] = *(const uint4*)(ap);                 // K khalf..khalf+7
        af.u[1] = *(const uint4*)(ap + 8);             // K khalf+8..khalf+15
        // B layout (contiguous K order, matching A): lanes 0-15 col=l, K=0..15;
        //                             lanes 16-31 col=l-16, K=16..31.
        F16 bf_[4];
        #pragma unroll
        for (int g = 0; g < 4; ++g) {
            const unsigned short* bp =
                &ldsB[buf][(g * 16 + (lane & 15)) * 32 + (lane >> 4) * 16];
            bf_[g].u[0] = *(const uint4*)(bp);
            bf_[g].u[1] = *(const uint4*)(bp + 8);
        }
        __builtin_amdgcn_sched_barrier(0);             // loads stay above
        #pragma unroll
        for (int g = 0; g < 4; ++g)
            acc[g] = __builtin_amdgcn_wmma_f32_16x16x32_bf16(
                false, af.v, false, bf_[g].v, (short)0, acc[g], false, false);
    };

    // prologue: stage K-step 0 into buffer 0
    {
        uint64_t ga = (uint64_t)(uintptr_t)(Bstage);
        asm volatile("global_load_async_to_lds_b128 %0, %1, off"
                     :: "v"(ldsDst), "v"(ga) : "memory");
    }

    int cur = 0;
    int kk = 0;
    for (; kk < K - 32; kk += 32) {                    // branch-free steady state
        {   // prefetch next K-step into the other buffer
            uint64_t ga = (uint64_t)(uintptr_t)(Bstage + kk + 32);
            uint32_t dst = ldsDst + (uint32_t)((cur ^ 1) * (64 * 32 * 2));
            asm volatile("global_load_async_to_lds_b128 %0, %1, off"
                         :: "v"(dst), "v"(ga) : "memory");
        }
        asm volatile("s_wait_asynccnt 1" ::: "memory");  // cur resident
        __syncthreads();
        compute(kk, cur);
        __syncthreads();                               // done reading cur
        cur ^= 1;
    }
    // peeled final K-step
    asm volatile("s_wait_asynccnt 0" ::: "memory");
    __syncthreads();
    compute(kk, cur);

    // C layout: lanes 0-15 N=l, VGPR r -> M=r; lanes 16-31 N=l-16, M=r+8.
    const int cr = (lane >> 4) * 8;
    const int cn = lane & 15;
    float* p = C + (size_t)(m0 + cr) * N + n0 + cn;    // one 64-bit mul total
    #pragma unroll
    for (int r = 0; r < 8; ++r) {
        #pragma unroll
        for (int g = 0; g < 4; ++g) p[g * 16] = acc[g][r];
        p += N;
    }
}

// -------- layer-1 aggregation: gather over CSR, ReLU, emit bf16 ------------
__global__ __launch_bounds__(256) void agg1_relu_bf16_kernel(
    const float* __restrict__ h, const int* __restrict__ rowptr,
    const int* __restrict__ esrc, const float* __restrict__ ewgt,
    unsigned short* __restrict__ outb) {
    const int n = blockIdx.x, b = blockIdx.y, tid = threadIdx.x;
    __shared__ int   ss[256];
    __shared__ float sw[256];
    const int e0 = rowptr[n], e1 = rowptr[n + 1];
    float acc[4] = {0.f, 0.f, 0.f, 0.f};
    for (int base = e0; base < e1; base += 256) {
        int cnt = e1 - base; if (cnt > 256) cnt = 256;
        if (tid < cnt) { ss[tid] = esrc[base + tid]; sw[tid] = ewgt[base + tid]; }
        __syncthreads();
        for (int j = 0; j < cnt; ++j) {
            const float w = sw[j];
            const float* hp = h + ((size_t)b * NN + ss[j]) * D1 + tid;
            acc[0] += w * hp[0];   acc[1] += w * hp[256];
            acc[2] += w * hp[512]; acc[3] += w * hp[768];
        }
        __syncthreads();
    }
    const size_t ob = ((size_t)b * NN + n) * D1 + tid;
    #pragma unroll
    for (int i = 0; i < 4; ++i) {
        float v = acc[i] > 0.f ? acc[i] : 0.f;
        outb[ob + (size_t)i * 256] = f2bf(v);
    }
}

// -------- layer-2 aggregation: gather over CSR, ReLU, emit fp32 ------------
__global__ __launch_bounds__(256) void agg2_relu_kernel(
    const float* __restrict__ h, const int* __restrict__ rowptr,
    const int* __restrict__ esrc, const float* __restrict__ ewgt,
    float* __restrict__ out) {
    const int n = blockIdx.x, b = blockIdx.y, tid = threadIdx.x;
    __shared__ int   ss[256];
    __shared__ float sw[256];
    const int e0 = rowptr[n], e1 = rowptr[n + 1];
    float acc = 0.f;
    for (int base = e0; base < e1; base += 256) {
        int cnt = e1 - base; if (cnt > 256) cnt = 256;
        if (tid < cnt) { ss[tid] = esrc[base + tid]; sw[tid] = ewgt[base + tid]; }
        __syncthreads();
        for (int j = 0; j < cnt; ++j)
            acc += sw[j] * h[((size_t)b * NN + ss[j]) * D2 + tid];
        __syncthreads();
    }
    out[((size_t)b * NN + n) * D2 + tid] = acc > 0.f ? acc : 0.f;
}

// ------------------- final projection (HBM-bound) --------------------------
__global__ void final_init_kernel(const float* __restrict__ bm,
                                  const float* __restrict__ bv,
                                  float* __restrict__ out) {
    int i = blockIdx.x * blockDim.x + threadIdx.x;    // 0..2047
    if (i < 1024) out[i] = bm[i & 127];
    else if (i < 2048) out[i] = bv[i & 127];
}

__global__ __launch_bounds__(256) void final_dot_kernel(
    const float* __restrict__ flat, const float* __restrict__ Wm,
    const float* __restrict__ Wv, float* __restrict__ out) {
    const int r = blockIdx.x;                          // 0..255 (128 mean +128 var)
    const int c = blockIdx.y;                          // 0..31 K-chunks
    const float* W = (r < 128 ? Wm : Wv) + (size_t)(r & 127) * KFLAT;
    const int ob = (r < 128 ? 0 : 1024) + (r & 127);
    const int k0 = c * (KFLAT / 32), k1 = k0 + (KFLAT / 32);   // 22480 each
    float acc[NB] = {};
    for (int k = k0 + threadIdx.x; k < k1; k += 256) {
        const float wv = W[k];                         // HBM stream (once)
        #pragma unroll
        for (int b = 0; b < NB; ++b)
            acc[b] += wv * flat[(size_t)b * KFLAT + k]; // L2-resident
    }
    __shared__ float red[256];
    for (int b = 0; b < NB; ++b) {
        red[threadIdx.x] = acc[b];
        __syncthreads();
        for (int s = 128; s > 0; s >>= 1) {
            if (threadIdx.x < s) red[threadIdx.x] += red[threadIdx.x + s];
            __syncthreads();
        }
        if (threadIdx.x == 0) atomicAdd(&out[ob + b * 128], red[0]);
        __syncthreads();
    }
}

// ---------------------------------------------------------------------------
extern "C" void kernel_launch(void* const* d_in, const int* in_sizes, int n_in,
                              void* d_out, int out_size, void* d_ws, size_t ws_size,
                              hipStream_t stream) {
    (void)in_sizes; (void)n_in; (void)out_size; (void)ws_size;
    const float* x  = (const float*)d_in[0];
    const int*   ei = (const int*)d_in[1];
    const float* ew = (const float*)d_in[2];
    const float* W1 = (const float*)d_in[3];
    const float* W2 = (const float*)d_in[4];
    const float* Wm = (const float*)d_in[5];
    const float* bm = (const float*)d_in[6];
    const float* Wv = (const float*)d_in[7];
    const float* bv = (const float*)d_in[8];
    float* out = (float*)d_out;

    // ---- workspace carve-up ----
    char* base = (char*)d_ws;
    size_t off = 0;
    auto carve = [&](size_t bytes) -> char* {
        char* p = base + off;
        off = (off + bytes + 255) & ~(size_t)255;
        return p;
    };
    unsigned short* xb   = (unsigned short*)carve((size_t)MROWS * D0 * 2);
    unsigned short* w1b  = (unsigned short*)carve((size_t)D1 * D0 * 2);
    unsigned short* w2b  = (unsigned short*)carve((size_t)D2 * D1 * 2);
    float*          h1r  = (float*)carve((size_t)MROWS * D1 * 4);
    unsigned short* h1b  = (unsigned short*)carve((size_t)MROWS * D1 * 2);
    float*          h2r  = (float*)carve((size_t)MROWS * D2 * 4);
    float*          h2   = (float*)carve((size_t)MROWS * D2 * 4);
    int*            deg  = (int*)carve((size_t)NN * 4);
    int*            rowp = (int*)carve((size_t)(NN + 1) * 4);
    int*            curs = (int*)carve((size_t)NN * 4);
    int*            esrc = (int*)carve((size_t)NE * 4);
    float*          ewo  = (float*)carve((size_t)NE * 4);

    // ---- bf16 conversion ----
    f32_to_bf16_kernel<<<2048, 256, 0, stream>>>(x,  xb,  MROWS * D0);
    f32_to_bf16_kernel<<<1024, 256, 0, stream>>>(W1, w1b, D1 * D0);
    f32_to_bf16_kernel<<<512,  256, 0, stream>>>(W2, w2b, D2 * D1);

    // ---- build CSR over dst ----
    zero_i32_kernel<<<(NN + 255) / 256, 256, 0, stream>>>(deg,  NN);
    zero_i32_kernel<<<(NN + 255) / 256, 256, 0, stream>>>(curs, NN);
    deg_kernel<<<(NE + 255) / 256, 256, 0, stream>>>(ei, deg);
    scan_kernel<<<1, 256, 0, stream>>>(deg, rowp);
    fill_kernel<<<(NE + 255) / 256, 256, 0, stream>>>(ei, ew, rowp, curs, esrc, ewo);

    // ---- layer 1: GEMM (bf16 WMMA) + gather/ReLU ----
    gemm_bf16_kernel<<<dim3((MROWS + 127) / 128, D1 / 64), 256, 0, stream>>>(
        xb, w1b, h1r, MROWS, D1, D0);
    agg1_relu_bf16_kernel<<<dim3(NN, NB), 256, 0, stream>>>(h1r, rowp, esrc, ewo, h1b);

    // ---- layer 2: GEMM (bf16 WMMA) + gather/ReLU ----
    gemm_bf16_kernel<<<dim3((MROWS + 127) / 128, D2 / 64), 256, 0, stream>>>(
        h1b, w2b, h2r, MROWS, D2, D1);
    agg2_relu_kernel<<<dim3(NN, NB), 256, 0, stream>>>(h2r, rowp, esrc, ewo, h2);

    // ---- final projection: mean / log_var ----
    final_init_kernel<<<8, 256, 0, stream>>>(bm, bv, out);
    final_dot_kernel<<<dim3(256, 32), 256, 0, stream>>>(h2, Wm, Wv, out);
}
